// CausalSelfAttention_42623255446168
// MI455X (gfx1250) — matrix operand order repaired and checked by reading, not compiled, not run on hardware
//
#include <hip/hip_runtime.h>
#include <hip/hip_bf16.h>
#include <math.h>

// Problem dims (fixed by the reference)
#define S_LEN 2048
#define DIM   1024
#define NH    16
#define KVH   4
#define HD    64
#define KVD   (KVH * HD)   // 256
#define HALF  32
#define PI_F  3.14159265358979323846f

typedef __attribute__((ext_vector_type(16))) __bf16 bf16x16;
typedef __attribute__((ext_vector_type(8)))  __bf16 bf16x8;
typedef __attribute__((ext_vector_type(8)))  float  f32x8;

union AFrag { bf16x16 v; bf16x8 h[2]; };

__device__ __forceinline__ f32x8 wmma_bf16(bf16x16 a, bf16x16 b, f32x8 c) {
    // D = A(16x32 bf16) * B(32x16 bf16) + C(16x16 f32)
    return __builtin_amdgcn_wmma_f32_16x16x32_bf16(
        /*neg_a=*/false, a, /*neg_b=*/false, b,
        /*c_mod=*/(short)0, c, /*reuse_a=*/false, /*reuse_b=*/false);
}

// ---------------------------------------------------------------------------
// Tensor Data Mover: 2-D tile DMA global -> LDS (CDNA5 TDM, TENSORcnt-tracked)
// D# built per CDNA5 ISA bit layout.
// ---------------------------------------------------------------------------
#if defined(__has_builtin)
#if __has_builtin(__builtin_amdgcn_tensor_load_to_lds)
#define HAVE_TDM 1
#endif
#endif

#ifdef HAVE_TDM
typedef __attribute__((ext_vector_type(4))) unsigned int u32x4;
typedef __attribute__((ext_vector_type(8))) int          i32x8;
typedef __attribute__((ext_vector_type(4))) int          i32x4;

__device__ __forceinline__ void tdm_load_tile(unsigned lds_byte_off,
                                              const void* gptr,
                                              unsigned tile_k,      // elems (dim0)
                                              unsigned tile_rows,   // rows  (dim1)
                                              unsigned tensor_k,    // elems
                                              unsigned tensor_rows, // rows
                                              unsigned row_stride)  // elems
{
    unsigned long long ga = (unsigned long long)(uintptr_t)gptr;
    u32x4 g0;
    g0[0] = 1u;                                   // count=1, user descriptor
    g0[1] = lds_byte_off;                         // lds_addr [63:32]
    g0[2] = (unsigned)ga;                         // global_addr [95:64]
    g0[3] = (unsigned)((ga >> 32) & 0x01ffffffu)  // global_addr [120:96]
            | 0x80000000u;                        // type=2 ("image") [127:126]
    i32x8 g1;
    g1[0] = 0x00010000;                                           // data_size=2B
    g1[1] = (int)((tensor_k & 0xffffu) << 16);                    // tensor_dim0 lo
    g1[2] = (int)((tensor_k >> 16) | ((tensor_rows & 0xffffu) << 16));
    g1[3] = (int)((tensor_rows >> 16) | ((tile_k & 0xffffu) << 16)); // tile_dim0
    g1[4] = (int)(tile_rows & 0xffffu);                           // tile_dim1
    g1[5] = (int)row_stride;                                      // dim0 stride lo
    g1[6] = 0;
    g1[7] = 0;
    i32x4 z4 = {0, 0, 0, 0};
#if __clang_major__ >= 23
    i32x8 z8 = {0, 0, 0, 0, 0, 0, 0, 0};
    __builtin_amdgcn_tensor_load_to_lds(g0, g1, z4, z4, z8, 0);
#else
    __builtin_amdgcn_tensor_load_to_lds(g0, g1, z4, z4, 0);
#endif
}
#endif // HAVE_TDM

// ---------------------------------------------------------------------------
// fp32 -> bf16 conversion
// ---------------------------------------------------------------------------
__global__ void f32_to_bf16_kernel(const float* __restrict__ in,
                                   __bf16* __restrict__ out, int n) {
    int i = blockIdx.x * blockDim.x + threadIdx.x;
    if (i < n) out[i] = (__bf16)in[i];
}

// ---------------------------------------------------------------------------
// Tiled WMMA GEMM:  C[M,N] = A[M,K] * B[N,K]^T   (A,B bf16 K-major, C f32)
// Block = 128x64 tile, 256 threads = 8 waves, each wave owns a 32x32 sub-tile
// (4 WMMAs per 32-deep K step, 2x fragment reuse). K-panels staged to LDS by
// the Tensor Data Mover (wave 0 issues, s_wait_tensorcnt, workgroup barrier);
// falls back to vector global->LDS staging if the TDM builtin is absent.
// Dynamic LDS so the D# lds_addr is a static byte offset: A @ 0, B @ 8192.
// ---------------------------------------------------------------------------
#define ALDS_BYTES (128 * 32 * 2)
#define BLDS_BYTES (64 * 32 * 2)

__global__ __launch_bounds__(256)
void gemm_bf16_wmma_kernel(const __bf16* __restrict__ A,
                           const __bf16* __restrict__ B,
                           float* __restrict__ C,
                           int M, int N, int K) {
    extern __shared__ __align__(32) __bf16 smem[];
    __bf16* As = smem;              // 128 x 32
    __bf16* Bs = smem + 128 * 32;   // 64 x 32

    const int t    = threadIdx.x;
    const int wave = t >> 5;
    const int lane = t & 31;
    const int lh   = lane >> 4;   // lane half (0/1)
    const int nn   = lane & 15;

    const int mb = blockIdx.y * 128;
    const int nb = blockIdx.x * 64;
    const int wm = (wave >> 1) * 32;  // wave row tile
    const int wn = (wave & 1) * 32;   // wave col tile

    f32x8 acc00 = {}, acc01 = {}, acc10 = {}, acc11 = {};

    for (int k0 = 0; k0 < K; k0 += 32) {
        __syncthreads();
#ifdef HAVE_TDM
        if (t < 32) {
            tdm_load_tile(0,          A + (size_t)mb * K + k0, 32, 128,
                          (unsigned)K, (unsigned)M, (unsigned)K);
            tdm_load_tile(ALDS_BYTES, B + (size_t)nb * K + k0, 32, 64,
                          (unsigned)K, (unsigned)N, (unsigned)K);
            __builtin_amdgcn_s_wait_tensorcnt(0);
        }
#else
        {   // A panel: 128x32, 2 threads/row, 16 elems each
            int ar = t >> 1, ac = (t & 1) * 16;
            *(bf16x8*)(As + ar * 32 + ac) =
                *(const bf16x8*)(A + (size_t)(mb + ar) * K + k0 + ac);
            *(bf16x8*)(As + ar * 32 + ac + 8) =
                *(const bf16x8*)(A + (size_t)(mb + ar) * K + k0 + ac + 8);
            // B panel: 64x32, 4 threads/row, 8 elems each
            int br = t >> 2, bc = (t & 3) * 8;
            *(bf16x8*)(Bs + br * 32 + bc) =
                *(const bf16x8*)(B + (size_t)(nb + br) * K + k0 + bc);
            if (k0 + 32 < K) {
                __builtin_prefetch(A + (size_t)(mb + ar) * K + k0 + 32 + ac, 0, 1);
                __builtin_prefetch(B + (size_t)(nb + br) * K + k0 + 32 + bc, 0, 1);
            }
        }
#endif
        __syncthreads();

        // A fragments (rows wm+nn and wm+16+nn); K range lh*16..+15
        AFrag a0, a1;
        a0.h[0] = *(const bf16x8*)(As + (wm + nn) * 32 + lh * 16);
        a0.h[1] = *(const bf16x8*)(As + (wm + nn) * 32 + lh * 16 + 8);
        a1.h[0] = *(const bf16x8*)(As + (wm + 16 + nn) * 32 + lh * 16);
        a1.h[1] = *(const bf16x8*)(As + (wm + 16 + nn) * 32 + lh * 16 + 8);
        // B fragments (cols wn+nn and wn+16+nn); K range lh*16..+15
        bf16x16 b0 = *(const bf16x16*)(Bs + (wn + nn) * 32 + lh * 16);
        bf16x16 b1 = *(const bf16x16*)(Bs + (wn + 16 + nn) * 32 + lh * 16);

        acc00 = wmma_bf16(a0.v, b0, acc00);
        acc01 = wmma_bf16(a0.v, b1, acc01);
        acc10 = wmma_bf16(a1.v, b0, acc10);
        acc11 = wmma_bf16(a1.v, b1, acc11);
    }

    for (int j = 0; j < 8; ++j) {
        int r0 = mb + wm + j + 8 * lh;
        int r1 = r0 + 16;
        C[(size_t)r0 * N + nb + wn + nn]      = acc00[j];
        C[(size_t)r0 * N + nb + wn + 16 + nn] = acc01[j];
        C[(size_t)r1 * N + nb + wn + nn]      = acc10[j];
        C[(size_t)r1 * N + nb + wn + 16 + nn] = acc11[j];
    }
}

// ---------------------------------------------------------------------------
// Q epilogue: RMS-norm (per 64-dim head) + polar rotary + gain.
// Emits bf16 sub-heads: qsub[(h*2+half)][s][32]
// ---------------------------------------------------------------------------
__global__ void q_post_kernel(const float* __restrict__ Qf,
                              const float* __restrict__ gain,
                              __bf16* __restrict__ qsub) {
    int idx = blockIdx.x * blockDim.x + threadIdx.x;   // 0 .. S*NH-1
    if (idx >= S_LEN * NH) return;
    int h = idx & (NH - 1);
    int s = idx >> 4;
    const float* q = Qf + (size_t)s * DIM + h * HD;

    float v[HD];
    float ss = 0.f;
    for (int d = 0; d < HD; ++d) { v[d] = q[d]; ss += v[d] * v[d]; }
    float r = rsqrtf(ss * (1.0f / HD) + 1.1920929e-7f);
    float g = gain[h];
    float radius = 1.0f / (1.0f + 0.01f * (float)s);

    __bf16* o1 = qsub + ((size_t)(h * 2 + 0) * S_LEN + s) * HALF;
    __bf16* o2 = qsub + ((size_t)(h * 2 + 1) * S_LEN + s) * HALF;
    for (int d = 0; d < HALF; ++d) {
        float ang = (float)s * (float)d * (PI_F / 32.0f);
        float c  = radius * __cosf(ang);
        float si = radius * __sinf(ang);
        float x1 = v[d] * r, x2 = v[d + HALF] * r;
        o1[d] = (__bf16)((x1 * c + x2 * si) * g);
        o2[d] = (__bf16)((-x1 * si + x2 * c) * g);
    }
}

// K epilogue: RMS-norm + polar rotary (no gain). ksub[(h*2+half)][s][32]
__global__ void k_post_kernel(const float* __restrict__ Kf,
                              __bf16* __restrict__ ksub) {
    int idx = blockIdx.x * blockDim.x + threadIdx.x;   // 0 .. S*KVH-1
    if (idx >= S_LEN * KVH) return;
    int h = idx & (KVH - 1);
    int s = idx >> 2;
    const float* k = Kf + (size_t)s * KVD + h * HD;

    float v[HD];
    float ss = 0.f;
    for (int d = 0; d < HD; ++d) { v[d] = k[d]; ss += v[d] * v[d]; }
    float r = rsqrtf(ss * (1.0f / HD) + 1.1920929e-7f);
    float radius = 1.0f / (1.0f + 0.01f * (float)s);

    __bf16* o1 = ksub + ((size_t)(h * 2 + 0) * S_LEN + s) * HALF;
    __bf16* o2 = ksub + ((size_t)(h * 2 + 1) * S_LEN + s) * HALF;
    for (int d = 0; d < HALF; ++d) {
        float ang = (float)s * (float)d * (PI_F / 32.0f);
        float c  = radius * __cosf(ang);
        float si = radius * __sinf(ang);
        float x1 = v[d] * r, x2 = v[d + HALF] * r;
        o1[d] = (__bf16)(x1 * c + x2 * si);
        o2[d] = (__bf16)(-x1 * si + x2 * c);
    }
}

// V epilogue: v *= sigmoid(g); split halves. vsub[(h*2+half)][s][32]
__global__ void v_post_kernel(const float* __restrict__ Vf,
                              const float* __restrict__ Gf,
                              __bf16* __restrict__ vsub) {
    int idx = blockIdx.x * blockDim.x + threadIdx.x;   // 0 .. S*KVH-1
    if (idx >= S_LEN * KVH) return;
    int h = idx & (KVH - 1);
    int s = idx >> 2;
    const float* vv = Vf + (size_t)s * KVD + h * HD;
    const float* gg = Gf + (size_t)s * KVD + h * HD;

    __bf16* o1 = vsub + ((size_t)(h * 2 + 0) * S_LEN + s) * HALF;
    __bf16* o2 = vsub + ((size_t)(h * 2 + 1) * S_LEN + s) * HALF;
    for (int d = 0; d < HALF; ++d) {
        float a = vv[d]        * (1.0f / (1.0f + __expf(-gg[d])));
        float b = vv[d + HALF] * (1.0f / (1.0f + __expf(-gg[d + HALF])));
        o1[d] = (__bf16)a;
        o2[d] = (__bf16)b;
    }
}

// ---------------------------------------------------------------------------
// Flash attention over 32 sub-heads of dim 32, causal, scale = 1/sqrt(32).
// One wave per (sub-head, 16-row query tile); streams 32-key blocks.
// S = Q*K^T via 2 WMMAs; online softmax with shfl_xor row reductions;
// P re-shaped through LDS (C-layout -> A-layout); P*V via 2 WMMAs.
// ---------------------------------------------------------------------------
__global__ __launch_bounds__(32)
void flash_attn_kernel(const __bf16* __restrict__ qsub,
                       const __bf16* __restrict__ ksub,
                       const __bf16* __restrict__ vsub,
                       float* __restrict__ aout) {
    __shared__ __align__(32) __bf16 lK[32 * 32];  // [key][dim]
    __shared__ __align__(32) __bf16 lV[32 * 32];  // [dim][key] (transposed)
    __shared__ __align__(32) __bf16 lP[16 * 32];  // [row][key]

    const int sh  = blockIdx.y;            // 0..31 sub-head
    const int qb  = blockIdx.x * 16;       // query tile base
    const int h   = sh >> 1;
    const int hf  = sh & 1;
    const int kvs = ((h >> 2) * 2) + hf;   // GQA: 4 Q-heads per KV-head

    const int lane = threadIdx.x;
    const int lh   = lane >> 4;
    const int nn   = lane & 15;
    const float scale = 0.17677669529663687f;  // 1/sqrt(32)

    // Q fragment (A layout), loaded once
    const __bf16* qp = qsub + ((size_t)sh * S_LEN + qb + nn) * HALF;
    AFrag qa;
    qa.h[0] = *(const bf16x8*)(qp + lh * 16);
    qa.h[1] = *(const bf16x8*)(qp + lh * 16 + 8);

    float m[8], l[8];
    f32x8 acc0 = {}, acc1 = {};
    for (int j = 0; j < 8; ++j) { m[j] = -3.0e38f; l[j] = 0.0f; }

    for (int kb = 0; kb <= qb + 15; kb += 32) {
        __syncthreads();
        // Stage K tile [key][dim] and V tile transposed [dim][key]
        const __bf16* kp = ksub + ((size_t)kvs * S_LEN + kb + lane) * HALF;
        *(bf16x16*)(lK + lane * 32) = *(const bf16x16*)kp;
        const __bf16* vp = vsub + ((size_t)kvs * S_LEN + kb + lane) * HALF;
        for (int d = 0; d < 32; ++d) lV[d * 32 + lane] = vp[d];
        __syncthreads();

        // S = Q * K^T  (B fragment: lane holds key n, dims lh*16..+15)
        bf16x16 kb0 = *(const bf16x16*)(lK + nn * 32 + lh * 16);
        bf16x16 kb1 = *(const bf16x16*)(lK + (16 + nn) * 32 + lh * 16);
        f32x8 z = {};
        f32x8 s0 = wmma_bf16(qa.v, kb0, z);
        f32x8 s1 = wmma_bf16(qa.v, kb1, z);

        // Online softmax (rows live across the 16-lane half-groups)
        for (int j = 0; j < 8; ++j) {
            int row  = qb + j + 8 * lh;
            float x0 = s0[j] * scale;
            float x1 = s1[j] * scale;
            if (kb + nn      > row) x0 = -3.0e38f;
            if (kb + 16 + nn > row) x1 = -3.0e38f;
            float mx = fmaxf(x0, x1);
            mx = fmaxf(mx, __shfl_xor(mx, 1, 32));
            mx = fmaxf(mx, __shfl_xor(mx, 2, 32));
            mx = fmaxf(mx, __shfl_xor(mx, 4, 32));
            mx = fmaxf(mx, __shfl_xor(mx, 8, 32));
            float mn = fmaxf(m[j], mx);
            float p0 = __expf(x0 - mn);
            float p1 = __expf(x1 - mn);
            float rs = p0 + p1;
            rs += __shfl_xor(rs, 1, 32);
            rs += __shfl_xor(rs, 2, 32);
            rs += __shfl_xor(rs, 4, 32);
            rs += __shfl_xor(rs, 8, 32);
            float alpha = __expf(m[j] - mn);
            l[j] = l[j] * alpha + rs;
            m[j] = mn;
            acc0[j] *= alpha;
            acc1[j] *= alpha;
            lP[(j + 8 * lh) * 32 + nn]      = (__bf16)p0;
            lP[(j + 8 * lh) * 32 + 16 + nn] = (__bf16)p1;
        }
        __syncthreads();

        // acc += P(16x32) * V(32x32): A-frag from lP, B-frags from lV
        AFrag pa;
        pa.h[0] = *(const bf16x8*)(lP + nn * 32 + lh * 16);
        pa.h[1] = *(const bf16x8*)(lP + nn * 32 + lh * 16 + 8);
        bf16x16 vb0 = *(const bf16x16*)(lV + nn * 32 + lh * 16);
        bf16x16 vb1 = *(const bf16x16*)(lV + (16 + nn) * 32 + lh * 16);
        acc0 = wmma_bf16(pa.v, vb0, acc0);
        acc1 = wmma_bf16(pa.v, vb1, acc1);
    }

    // Normalize and store: aout[sh][row][32]
    for (int j = 0; j < 8; ++j) {
        float inv = 1.0f / l[j];
        size_t base = ((size_t)sh * S_LEN + qb + j + 8 * lh) * HALF;
        aout[base + nn]      = acc0[j] * inv;
        aout[base + 16 + nn] = acc1[j] * inv;
    }
}

// ---------------------------------------------------------------------------
// y = concat(a1 - lambda*a2, a1 + lambda*a2) per head, emitted as bf16 [s][1024]
// ---------------------------------------------------------------------------
__global__ void combine_kernel(const float* __restrict__ aout,
                               const float* __restrict__ lambda_p,
                               __bf16* __restrict__ yb) {
    int idx = blockIdx.x * blockDim.x + threadIdx.x;   // 0 .. S*DIM-1
    if (idx >= S_LEN * DIM) return;
    int s = idx >> 10;
    int c = idx & (DIM - 1);
    int h = c >> 6;
    int d = c & (HD - 1);
    int dd = d & (HALF - 1);
    float lam = lambda_p[h];
    float a1 = aout[((size_t)(h * 2 + 0) * S_LEN + s) * HALF + dd];
    float a2 = aout[((size_t)(h * 2 + 1) * S_LEN + s) * HALF + dd];
    float y = (d < HALF) ? (a1 - lam * a2) : (a1 + lam * a2);
    yb[idx] = (__bf16)y;
}

// ---------------------------------------------------------------------------
// Host-side launcher
// ---------------------------------------------------------------------------
extern "C" void kernel_launch(void* const* d_in, const int* in_sizes, int n_in,
                              void* d_out, int out_size, void* d_ws, size_t ws_size,
                              hipStream_t stream) {
    const float* x   = (const float*)d_in[0];
    const float* Wq  = (const float*)d_in[1];
    const float* Wk  = (const float*)d_in[2];
    const float* Wv  = (const float*)d_in[3];
    const float* Wg  = (const float*)d_in[4];
    const float* Wo  = (const float*)d_in[5];
    const float* qg  = (const float*)d_in[6];
    const float* lam = (const float*)d_in[7];
    float* out = (float*)d_out;

    char* ws = (char*)d_ws;
    size_t off = 0;
    auto carve = [&](size_t bytes) -> char* {
        char* p = ws + off;
        off += (bytes + 255) & ~(size_t)255;
        return p;
    };

    __bf16* xb   = (__bf16*)carve((size_t)S_LEN * DIM * 2);
    __bf16* wqb  = (__bf16*)carve((size_t)DIM * DIM * 2);
    __bf16* wkb  = (__bf16*)carve((size_t)KVD * DIM * 2);
    __bf16* wvb  = (__bf16*)carve((size_t)KVD * DIM * 2);
    __bf16* wgb  = (__bf16*)carve((size_t)KVD * DIM * 2);
    __bf16* wob  = (__bf16*)carve((size_t)DIM * DIM * 2);
    float*  Qf   = (float*)carve((size_t)S_LEN * DIM * 4);
    float*  Kf   = (float*)carve((size_t)S_LEN * KVD * 4);
    float*  Vf   = (float*)carve((size_t)S_LEN * KVD * 4);
    float*  Gf   = (float*)carve((size_t)S_LEN * KVD * 4);
    __bf16* qsub = (__bf16*)carve((size_t)2 * NH  * S_LEN * HALF * 2);
    __bf16* ksub = (__bf16*)carve((size_t)2 * KVH * S_LEN * HALF * 2);
    __bf16* vsub = (__bf16*)carve((size_t)2 * KVH * S_LEN * HALF * 2);
    float*  aout = (float*)carve((size_t)2 * NH * S_LEN * HALF * 4);
    __bf16* yb   = (__bf16*)carve((size_t)S_LEN * DIM * 2);
    (void)ws_size; (void)in_sizes; (void)n_in; (void)out_size;

    auto cvt = [&](const float* src, __bf16* dst, int n) {
        f32_to_bf16_kernel<<<(n + 255) / 256, 256, 0, stream>>>(src, dst, n);
    };
    cvt(x,  xb,  S_LEN * DIM);
    cvt(Wq, wqb, DIM * DIM);
    cvt(Wk, wkb, KVD * DIM);
    cvt(Wv, wvb, KVD * DIM);
    cvt(Wg, wgb, KVD * DIM);
    cvt(Wo, wob, DIM * DIM);

    const size_t gemm_lds = (size_t)ALDS_BYTES + BLDS_BYTES;  // 12 KB

    // Projections: C = x * W^T  (both K-major)
    gemm_bf16_wmma_kernel<<<dim3(DIM / 64, S_LEN / 128), 256, gemm_lds, stream>>>(
        xb, wqb, Qf, S_LEN, DIM, DIM);
    gemm_bf16_wmma_kernel<<<dim3(KVD / 64, S_LEN / 128), 256, gemm_lds, stream>>>(
        xb, wkb, Kf, S_LEN, KVD, DIM);
    gemm_bf16_wmma_kernel<<<dim3(KVD / 64, S_LEN / 128), 256, gemm_lds, stream>>>(
        xb, wvb, Vf, S_LEN, KVD, DIM);
    gemm_bf16_wmma_kernel<<<dim3(KVD / 64, S_LEN / 128), 256, gemm_lds, stream>>>(
        xb, wgb, Gf, S_LEN, KVD, DIM);

    q_post_kernel<<<(S_LEN * NH + 255) / 256, 256, 0, stream>>>(Qf, qg, qsub);
    k_post_kernel<<<(S_LEN * KVH + 255) / 256, 256, 0, stream>>>(Kf, ksub);
    v_post_kernel<<<(S_LEN * KVH + 255) / 256, 256, 0, stream>>>(Vf, Gf, vsub);

    flash_attn_kernel<<<dim3(S_LEN / 16, 2 * NH), 32, 0, stream>>>(
        qsub, ksub, vsub, aout);

    combine_kernel<<<(S_LEN * DIM + 255) / 256, 256, 0, stream>>>(aout, lam, yb);

    // Output projection -> d_out (f32)
    gemm_bf16_wmma_kernel<<<dim3(DIM / 64, S_LEN / 128), 256, gemm_lds, stream>>>(
        yb, wob, out, S_LEN, DIM, DIM);
}
